// StateTracker_SASRec_7559142441436
// MI455X (gfx1250) — matrix operand order repaired and checked by reading, not compiled, not run on hardware
//
#include <hip/hip_runtime.h>
#include <hip/hip_bf16.h>

// ---------------------------------------------------------------------------
// SASRec final-state extractor for MI455X (gfx1250, wave32, WMMA).
// One workgroup per batch row. K/V projections (the only full-sequence GEMMs)
// run on v_wmma_f32_16x16x32_bf16 with b128 LDS fragment loads; everything
// else is single-row VALU work.
// ---------------------------------------------------------------------------

typedef __attribute__((ext_vector_type(16))) __bf16 v16bf;
typedef __attribute__((ext_vector_type(8)))  float  v8f;

#define HD   128          // hidden size
#define WSEQ 200          // window size
#define WPAD 208          // 13 * 16 (WMMA M tiles)
#define MTILES 13
#define NEGV (-4294967295.0f)
#define LNEPS 1e-5f

// ---- LDS layout (bytes). Total 230,848 B < 320 KB/WGP. All 16B aligned. ----
#define OFF_SEQ   0               // ushort[WPAD*HD]  bf16 seq  [t][h]   (53248)
#define OFF_KB    53248           // ushort[WPAD*HD]  bf16 K    [t][h]   (53248)
#define OFF_VT    106496          // ushort[HD*WPAD]  bf16 V^T  [h][t]   (53248)
#define OFF_WK    159744          // ushort[HD*HD]    bf16 wk^T [n][k]   (32768)
#define OFF_WV    192512          // ushort[HD*HD]    bf16 wv^T [n][k]   (32768)
#define OFF_SC    225280          // float[2*WPAD]    scores/attn        (1664)
#define OFF_KM    226944          // float[WPAD]      key mask           (832)
#define OFF_Q     227776          // float[HD]        queries row
#define OFF_QP    228288          // float[HD]        Q projection row
#define OFF_FF    228800          // float[HD]        ff_in row
#define OFF_Y1    229312          // float[HD]        relu hidden row
#define OFF_RED   229824          // float[256]       reduction scratch
#define SMEM_BYTES 230848

__device__ __forceinline__ unsigned short f2us_bf(float f) {
  unsigned u = __float_as_uint(f);
  u = (u + 0x7FFFu + ((u >> 16) & 1u)) >> 16;   // round-to-nearest-even
  return (unsigned short)u;
}
__device__ __forceinline__ float bfLo(unsigned u) {
  return __uint_as_float(u << 16);
}
__device__ __forceinline__ float bfHi(unsigned u) {
  return __uint_as_float(u & 0xffff0000u);
}
__device__ __forceinline__ unsigned packbf(float a, float b) {
  return (unsigned)f2us_bf(a) | ((unsigned)f2us_bf(b) << 16);
}

union BfFrag { v16bf v; uint4 u[2]; };

__device__ __forceinline__ v8f wmma_bf16(v16bf a, v16bf b, v8f c) {
  // (neg_a, A, neg_b, B, c_mod, C, reuse_a, reuse_b)
  return __builtin_amdgcn_wmma_f32_16x16x32_bf16(false, a, false, b, (short)0, c,
                                                 false, false);
}

// A-fragment 16x32 bf16 (ISA 7.12.2): lane<16 -> K={0..7,16..23}, lane>=16 ->
// +8. Both halves are contiguous 8-element runs -> two ds_load_b128.
__device__ __forceinline__ v16bf loadA(const unsigned short* sSeq, int mt, int kk,
                                       int lane) {
  int m = lane & 15;
  int khi = (lane >> 4) << 3;                       // 0 or 8
  const unsigned short* src = sSeq + (mt * 16 + m) * HD + kk * 32 + khi;
  BfFrag f;
  f.u[0] = *(const uint4*)(src);
  f.u[1] = *(const uint4*)(src + 16);
  return f.v;
}

// B-fragment 32x16 bf16 from a TRANSPOSED weight [n][k]: lane holds column
// n = lane&15, K={0..7,16..23} (+8 for lane>=16) -> two ds_load_b128.
__device__ __forceinline__ v16bf loadBT(const unsigned short* sWT, int kk, int nt,
                                        int lane) {
  int n = lane & 15;
  int khi = (lane >> 4) << 3;                       // 0 or 8
  const unsigned short* src = sWT + (nt * 16 + n) * HD + kk * 32 + khi;
  BfFrag f;
  f.u[0] = *(const uint4*)(src);
  f.u[1] = *(const uint4*)(src + 16);
  return f.v;
}

__device__ __forceinline__ float blockSum(float v, float* sRed) {
  int t = threadIdx.x;
  sRed[t] = v;
  __syncthreads();
#pragma unroll
  for (int s = 128; s > 0; s >>= 1) {
    if (t < s) sRed[t] += sRed[t + s];
    __syncthreads();
  }
  float r = sRed[0];
  __syncthreads();
  return r;
}

// LayerNorm over a 128-vector distributed one element per thread (j<128).
__device__ __forceinline__ float doLN(float x, bool act, const float* g,
                                      const float* b, int j, float* sRed) {
  float m = blockSum(act ? x : 0.f, sRed) * (1.f / 128.f);
  float d = x - m;
  float var = blockSum(act ? d * d : 0.f, sRed) * (1.f / 128.f);
  float r = __frsqrt_rn(var + LNEPS);
  return act ? (d * r * g[j] + b[j]) : 0.f;
}

__global__ __launch_bounds__(256)
void sasrec_state_kernel(const int* __restrict__ items,
                         const int* __restrict__ lens,
                         const float* __restrict__ emb,
                         const float* __restrict__ pos,
                         const float* __restrict__ wq, const float* __restrict__ bq,
                         const float* __restrict__ wk, const float* __restrict__ bk,
                         const float* __restrict__ wv, const float* __restrict__ bv,
                         const float* __restrict__ w1, const float* __restrict__ b1,
                         const float* __restrict__ w2, const float* __restrict__ b2,
                         const float* __restrict__ ln1g, const float* __restrict__ ln1b,
                         const float* __restrict__ ln2g, const float* __restrict__ ln2b,
                         const float* __restrict__ ln3g, const float* __restrict__ ln3b,
                         const float* __restrict__ ffg,  const float* __restrict__ ffb,
                         float* __restrict__ out) {
  extern __shared__ char smem[];
  unsigned short* sSeq = (unsigned short*)(smem + OFF_SEQ);
  unsigned short* sKb  = (unsigned short*)(smem + OFF_KB);
  unsigned short* sVT  = (unsigned short*)(smem + OFF_VT);
  unsigned short* sWkT = (unsigned short*)(smem + OFF_WK);
  unsigned short* sWvT = (unsigned short*)(smem + OFF_WV);
  float* sScore = (float*)(smem + OFF_SC);
  float* sKm    = (float*)(smem + OFF_KM);
  float* sQ     = (float*)(smem + OFF_Q);
  float* sQp    = (float*)(smem + OFF_QP);
  float* sFf    = (float*)(smem + OFF_FF);
  float* sY1    = (float*)(smem + OFF_Y1);
  float* sRed   = (float*)(smem + OFF_RED);

  const int b    = blockIdx.x;
  const int tid  = threadIdx.x;
  const int lane = tid & 31;
  const int wave = tid >> 5;
  const float sqrtH = 11.313708498984761f;       // sqrt(128)
  const float invSqrtH = 0.08838834764831845f;   // 1/sqrt(128)

  const int len = lens[b];
  const int idx = min(max(len - 1, 0), WSEQ - 1);
  const float maskv = (len > 0) ? 1.f : 0.f;

  // --- Stage bf16 weights TRANSPOSED [n][k] (L2-hot across all blocks) ---
  for (int e = tid; e < HD * HD / 2; e += 256) {
    int n = e >> 6, k = (e & 63) << 1;
    ((unsigned*)sWkT)[e] = packbf(wk[k * HD + n], wk[(k + 1) * HD + n]);
    ((unsigned*)sWvT)[e] = packbf(wv[k * HD + n], wv[(k + 1) * HD + n]);
  }
  // --- Stage seq = emb[items]*mask*sqrt(H) + pos (bf16 pairs, 208 rows) ---
  for (int e = tid; e < WPAD * HD / 2; e += 256) {
    int t = e >> 6, h = (e & 63) << 1;
    float f0 = 0.f, f1 = 0.f;
    if (t < WSEQ) {
      if (t < len) {
        long base = (long)items[b * WSEQ + t] * HD + h;
        f0 = emb[base] * sqrtH;
        f1 = emb[base + 1] * sqrtH;
      }
      f0 += pos[t * HD + h];
      f1 += pos[t * HD + h + 1];
    }
    ((unsigned*)sSeq)[e] = packbf(f0, f1);
  }
  __syncthreads();

  // --- key_mask[t] = sign(|sum(seq[t])|), vectorized b32 reads ---
  if (tid < WPAD) {
    const unsigned* sr = (const unsigned*)(sSeq + tid * HD);
    float s = 0.f;
#pragma unroll 8
    for (int c = 0; c < HD / 2; ++c) { unsigned u = sr[c]; s += bfLo(u) + bfHi(u); }
    sKm[tid] = (s != 0.f) ? 1.f : 0.f;
  }

  // --- K = seq@wk + bk, V = seq@wv + bv via WMMA (each wave owns M tiles) ---
  for (int mt = wave; mt < MTILES; mt += 8) {
    v16bf a[4];
#pragma unroll
    for (int kkk = 0; kkk < 4; ++kkk) a[kkk] = loadA(sSeq, mt, kkk, lane);
    for (int nt = 0; nt < 8; ++nt) {
      v8f accK = {}; v8f accV = {};
#pragma unroll
      for (int kkk = 0; kkk < 4; ++kkk) {
        v16bf fbK = loadBT(sWkT, kkk, nt, lane);
        v16bf fbV = loadBT(sWvT, kkk, nt, lane);
        accK = wmma_bf16(a[kkk], fbK, accK);
        accV = wmma_bf16(a[kkk], fbV, accV);
      }
      int col = lane & 15;
      int rbase = (lane >> 4) << 3;                 // D rows r -> t = base + r
      int n = nt * 16 + col;
      int tbase = mt * 16 + rbase;
      float bK = bk[n], bV = bv[n];
#pragma unroll
      for (int r = 0; r < 8; ++r)
        sKb[(tbase + r) * HD + n] = f2us_bf(accK[r] + bK);
      uint4 vp;                                     // 8 consecutive t -> b128
      vp.x = packbf(accV[0] + bV, accV[1] + bV);
      vp.y = packbf(accV[2] + bV, accV[3] + bV);
      vp.z = packbf(accV[4] + bV, accV[5] + bV);
      vp.w = packbf(accV[6] + bV, accV[7] + bV);
      *(uint4*)(sVT + n * WPAD + tbase) = vp;
    }
  }
  __syncthreads();

  // --- queries row at idx in f32: LN1(seq[idx]) ---
  const int j = tid;            // element owner for j<128
  float x = 0.f;
  if (j < HD) {
    if (idx < len) x = emb[(long)items[b * WSEQ + idx] * HD + j] * sqrtH;
    x += pos[idx * HD + j];
  }
  float q = doLN(x, j < HD, ln1g, ln1b, j, sRed);
  if (j < HD) sQ[j] = q;
  float qsum = blockSum(j < HD ? q : 0.f, sRed);
  float qmask = (qsum != 0.f) ? 1.f : 0.f;

  // --- Q = queries @ wq + bq (one row) ---
  if (j < HD) {
    float acc = bq[j];
    for (int i = 0; i < HD; ++i) acc = fmaf(sQ[i], wq[i * HD + j], acc);
    sQp[j] = acc;
  }
  __syncthreads();

  // --- scores (one query row, 2 heads, causal + key mask), b128 K reads ---
  for (int e = tid; e < 2 * WPAD; e += 256) {
    int hh = e / WPAD, k = e - hh * WPAD;
    float sc = NEGV;
    if (k < WSEQ && k <= idx && sKm[k] != 0.f) {
      const uint4*  kr = (const uint4*)(sKb + k * HD + hh * 64);
      const float4* qq = (const float4*)(sQp + hh * 64);
      float dot = 0.f;
#pragma unroll
      for (int c = 0; c < 8; ++c) {
        uint4 u = kr[c];
        float4 q0 = qq[2 * c], q1 = qq[2 * c + 1];
        dot = fmaf(q0.x, bfLo(u.x), dot); dot = fmaf(q0.y, bfHi(u.x), dot);
        dot = fmaf(q0.z, bfLo(u.y), dot); dot = fmaf(q0.w, bfHi(u.y), dot);
        dot = fmaf(q1.x, bfLo(u.z), dot); dot = fmaf(q1.y, bfHi(u.z), dot);
        dot = fmaf(q1.z, bfLo(u.w), dot); dot = fmaf(q1.w, bfHi(u.w), dot);
      }
      sc = dot * invSqrtH;
    }
    sScore[e] = sc;
  }
  __syncthreads();

  // --- softmax per head: wave 0 -> head 0, wave 1 -> head 1 (wave32 shfl) ---
  if (wave < 2) {
    float* sc = sScore + wave * WPAD;
    float m = -3.4e38f;
    for (int k = lane; k < WSEQ; k += 32) m = fmaxf(m, sc[k]);
#pragma unroll
    for (int off = 16; off; off >>= 1) m = fmaxf(m, __shfl_xor(m, off, 32));
    float s = 0.f;
    for (int k = lane; k < WSEQ; k += 32) s += __expf(sc[k] - m);
#pragma unroll
    for (int off = 16; off; off >>= 1) s += __shfl_xor(s, off, 32);
    float inv = qmask / s;
    for (int k = lane; k < WSEQ; k += 32) sc[k] = __expf(sc[k] - m) * inv;
    for (int k = WSEQ + lane; k < WPAD; k += 32) sc[k] = 0.f;
  }
  __syncthreads();

  // --- ctx + residual (V^T row is contiguous -> b128 reads) ---
  float mh = 0.f;
  if (j < HD) {
    const uint4*  vr = (const uint4*)(sVT + j * WPAD);
    const float4* at = (const float4*)(sScore + (j >> 6) * WPAD);
    float c = 0.f;
#pragma unroll 5
    for (int cc = 0; cc < WSEQ / 8; ++cc) {         // 200 = 25 * 8 exactly
      uint4 u = vr[cc];
      float4 a0 = at[2 * cc], a1 = at[2 * cc + 1];
      c = fmaf(a0.x, bfLo(u.x), c); c = fmaf(a0.y, bfHi(u.x), c);
      c = fmaf(a0.z, bfLo(u.y), c); c = fmaf(a0.w, bfHi(u.y), c);
      c = fmaf(a1.x, bfLo(u.z), c); c = fmaf(a1.y, bfHi(u.z), c);
      c = fmaf(a1.z, bfLo(u.w), c); c = fmaf(a1.w, bfHi(u.w), c);
    }
    mh = c + sQ[j];
  }
  float ffin = doLN(mh, j < HD, ln2g, ln2b, j, sRed);
  if (j < HD) sFf[j] = ffin;
  __syncthreads();

  if (j < HD) {
    float acc = b1[j];
    for (int i = 0; i < HD; ++i) acc = fmaf(sFf[i], w1[i * HD + j], acc);
    sY1[j] = fmaxf(acc, 0.f);
  }
  __syncthreads();

  float z = 0.f;
  if (j < HD) {
    float acc = b2[j];
    for (int i = 0; i < HD; ++i) acc = fmaf(sY1[i], w2[i * HD + j], acc);
    z = acc + sFf[j];
  }
  float ff = doLN(z, j < HD, ffg, ffb, j, sRed);
  ff *= maskv;
  float res = doLN(ff, j < HD, ln3g, ln3b, j, sRed);
  if (j < HD) out[b * HD + j] = res;
}

extern "C" void kernel_launch(void* const* d_in, const int* in_sizes, int n_in,
                              void* d_out, int out_size, void* d_ws, size_t ws_size,
                              hipStream_t stream) {
  (void)n_in; (void)d_ws; (void)ws_size; (void)out_size;
  const int*   items = (const int*)d_in[0];
  const int*   lens  = (const int*)d_in[1];
  const float* emb   = (const float*)d_in[2];
  const float* pos   = (const float*)d_in[3];
  const float* wq    = (const float*)d_in[4];
  const float* bq    = (const float*)d_in[5];
  const float* wk    = (const float*)d_in[6];
  const float* bk    = (const float*)d_in[7];
  const float* wv    = (const float*)d_in[8];
  const float* bv    = (const float*)d_in[9];
  const float* w1    = (const float*)d_in[10];
  const float* b1    = (const float*)d_in[11];
  const float* w2    = (const float*)d_in[12];
  const float* b2    = (const float*)d_in[13];
  const float* ln1g  = (const float*)d_in[14];
  const float* ln1b  = (const float*)d_in[15];
  const float* ln2g  = (const float*)d_in[16];
  const float* ln2b  = (const float*)d_in[17];
  const float* ln3g  = (const float*)d_in[18];
  const float* ln3b  = (const float*)d_in[19];
  const float* ffg   = (const float*)d_in[20];
  const float* ffb   = (const float*)d_in[21];
  float* out = (float*)d_out;

  const int B = in_sizes[1];   // len_states count == batch
  sasrec_state_kernel<<<dim3(B), dim3(256), SMEM_BYTES, stream>>>(
      items, lens, emb, pos, wq, bq, wk, bk, wv, bv, w1, b1, w2, b2,
      ln1g, ln1b, ln2g, ln2b, ln3g, ln3b, ffg, ffb, out);
}
